// PosNegCrossAttention_89412629168756
// MI455X (gfx1250) — matrix operand order repaired and checked by reading, not compiled, not run on hardware
//
#include <hip/hip_runtime.h>
#include <hip/hip_bf16.h>

// ---------------------------------------------------------------------------
// PosNegCrossAttention for MI455X (gfx1250), wave32, WMMA f32_16x16x32_f16.
// B=4, LQ=LK=1024, EMBED=1024, H=16, Dh=64.
//
// d_out (floats): pos_output[4M] | neg_output[4M] | pos_attn[64M] | neg_attn[64M]
// d_ws  (f16):    Xq(8M) Xk(8M) Xv(8M) Wq(2M) Wk(2M) Wv(2M) Wo(2M)
//                 Qh(8M) Kh(8M) Vt(8M) Cp(8M) Cn(8M)   -> 72 MB total
//
// All GEMMs: one wave owns a 16(M) x 64(N) output strip -> 4 f32 accumulators,
// A-tile loaded once per k-step, reused by 4 WMMAs. K-loops are software-
// pipelined (next step's tiles loaded during current step's WMMAs) so the
// per-WMMA s_wait_loadcnt 0x0 serialization disappears.
// ---------------------------------------------------------------------------

typedef __attribute__((ext_vector_type(16))) _Float16 v16h;
typedef __attribute__((ext_vector_type(8)))  _Float16 v8h;
typedef __attribute__((ext_vector_type(4)))  _Float16 v4h;
typedef __attribute__((ext_vector_type(8)))  float    v8f;
typedef __attribute__((ext_vector_type(4)))  float    v4f;

#define WMMA_F16(a, b, c) \
  __builtin_amdgcn_wmma_f32_16x16x32_f16(false, (a), false, (b), (short)0, (c), false, false)

// ---- per-lane tile loaders (wave32 WMMA register layouts, ISA 7.12.2) ------

// A-matrix 16x32 f16: lane<16 -> row=lane, K=[0..7],[16..23];
//                     lane>=16 -> row=lane-16, K=[8..15],[24..31]
__device__ __forceinline__ v16h load_a_f16(const _Float16* __restrict__ p, int ld, int lane) {
  const _Float16* r0 = p + (lane & 15) * ld + ((lane >> 4) << 3);
  v8h lo = *(const v8h*)(r0);
  v8h hi = *(const v8h*)(r0 + 16);
  return __builtin_shufflevector(lo, hi, 0, 1, 2, 3, 4, 5, 6, 7,
                                 8, 9, 10, 11, 12, 13, 14, 15);
}

// A-tile with f32 source (attention probabilities), converted in-register.
__device__ __forceinline__ v16h load_a_f32(const float* __restrict__ p, int ld, int lane) {
  const float* r0 = p + (lane & 15) * ld + ((lane >> 4) << 3);
  v16h a;
#pragma unroll
  for (int i = 0; i < 8; ++i) {
    a[i]     = (_Float16)r0[i];
    a[i + 8] = (_Float16)r0[i + 16];
  }
  return a;
}

// B-matrix 32x16 f16 (KxN) from row-major [N x K] source (implicit W^T):
// lanes 0-15: col=lane, K=[0..7],[16..23]; lanes 16-31: col=lane-16, K=[8..15],[24..31]
__device__ __forceinline__ v16h load_b_f16(const _Float16* __restrict__ p, int ld, int lane) {
  const _Float16* r0 = p + (lane & 15) * ld + ((lane >> 4) << 3);
  v8h lo = *(const v8h*)(r0);
  v8h hi = *(const v8h*)(r0 + 16);
  return __builtin_shufflevector(lo, hi, 0, 1, 2, 3, 4, 5, 6, 7,
                                 8, 9, 10, 11, 12, 13, 14, 15);
}

// ---------------------------------------------------------------------------
// K0: bulk f32 -> f16 conversion (one-time; X and W streams)
// ---------------------------------------------------------------------------
__global__ __launch_bounds__(256) void cvt_f32_f16(const float* __restrict__ src,
                                                   _Float16* __restrict__ dst, int n4) {
  const int i = blockIdx.x * 256 + threadIdx.x;
  if (i >= n4) return;
  v4f v = ((const v4f*)src)[i];
  v4h h;
#pragma unroll
  for (int j = 0; j < 4; ++j) h[j] = (_Float16)v[j];
  ((v4h*)dst)[i] = h;
}

// ---------------------------------------------------------------------------
// K1: Y[m,n] = sum_k X[m,k]*W[n,k] + bias[n]; X,W f16; out f16 flat [4096x1024].
// grid(256,4) block(32,4): wave = 16x64 strip; ng = by*4+ty in 0..15.
// ---------------------------------------------------------------------------
__global__ __launch_bounds__(128) void proj_xwT(const _Float16* __restrict__ X,
                                                const _Float16* __restrict__ W,
                                                const float* __restrict__ bias,
                                                _Float16* __restrict__ out) {
  const int lane = threadIdx.x;
  const int mt = blockIdx.x;                               // 0..255
  const int ng = blockIdx.y * blockDim.y + threadIdx.y;    // 0..15
  const _Float16* Ab = X + (size_t)(mt * 16) * 1024;
  const _Float16* Bb = W + (size_t)(ng * 64) * 1024;
  v8f c0 = {}, c1 = {}, c2 = {}, c3 = {};

  v16h a  = load_a_f16(Ab, 1024, lane);
  v16h b0 = load_b_f16(Bb,             1024, lane);
  v16h b1 = load_b_f16(Bb + 16 * 1024, 1024, lane);
  v16h b2 = load_b_f16(Bb + 32 * 1024, 1024, lane);
  v16h b3 = load_b_f16(Bb + 48 * 1024, 1024, lane);
  for (int k0 = 32; k0 < 1024; k0 += 32) {
    v16h an  = load_a_f16(Ab + k0, 1024, lane);
    v16h bn0 = load_b_f16(Bb + k0,             1024, lane);
    v16h bn1 = load_b_f16(Bb + k0 + 16 * 1024, 1024, lane);
    v16h bn2 = load_b_f16(Bb + k0 + 32 * 1024, 1024, lane);
    v16h bn3 = load_b_f16(Bb + k0 + 48 * 1024, 1024, lane);
    c0 = WMMA_F16(a, b0, c0);
    c1 = WMMA_F16(a, b1, c1);
    c2 = WMMA_F16(a, b2, c2);
    c3 = WMMA_F16(a, b3, c3);
    a = an; b0 = bn0; b1 = bn1; b2 = bn2; b3 = bn3;
  }
  c0 = WMMA_F16(a, b0, c0);
  c1 = WMMA_F16(a, b1, c1);
  c2 = WMMA_F16(a, b2, c2);
  c3 = WMMA_F16(a, b3, c3);

  const int m0 = mt * 16 + ((lane >> 4) << 3);
  const int nb = ng * 64 + (lane & 15);
  _Float16* o = out + (size_t)m0 * 1024 + nb;
  v8f cc[4] = {c0, c1, c2, c3};
#pragma unroll
  for (int j = 0; j < 4; ++j) {
    const float bn = bias[nb + j * 16];
#pragma unroll
    for (int r = 0; r < 8; ++r)
      o[(size_t)r * 1024 + j * 16] = (_Float16)(cc[j][r] + bn);
  }
}

// ---------------------------------------------------------------------------
// K2: transposed V projection. Vt[b, m, kpos] = sum_k Wv[m,k]*X[b*1024+kpos,k]+bv[m]
// A = Wv rows (m=dim), B = X rows (n=token) -> D already transposed.
// grid(64,16) block(32,4): mt=dim tile, tg = by*4+ty in 0..63 (4 token tiles each).
// ---------------------------------------------------------------------------
__global__ __launch_bounds__(128) void proj_v_t(const _Float16* __restrict__ X,
                                                const _Float16* __restrict__ W,
                                                const float* __restrict__ bias,
                                                _Float16* __restrict__ vt) {
  const int lane = threadIdx.x;
  const int mt = blockIdx.x;                               // 0..63 (dim tiles)
  const int tg = blockIdx.y * blockDim.y + threadIdx.y;    // 0..63 (token groups)
  const _Float16* Ab = W + (size_t)(mt * 16) * 1024;
  const _Float16* Bb = X + (size_t)(tg * 64) * 1024;
  v8f c0 = {}, c1 = {}, c2 = {}, c3 = {};

  v16h a  = load_a_f16(Ab, 1024, lane);
  v16h b0 = load_b_f16(Bb,             1024, lane);
  v16h b1 = load_b_f16(Bb + 16 * 1024, 1024, lane);
  v16h b2 = load_b_f16(Bb + 32 * 1024, 1024, lane);
  v16h b3 = load_b_f16(Bb + 48 * 1024, 1024, lane);
  for (int k0 = 32; k0 < 1024; k0 += 32) {
    v16h an  = load_a_f16(Ab + k0, 1024, lane);
    v16h bn0 = load_b_f16(Bb + k0,             1024, lane);
    v16h bn1 = load_b_f16(Bb + k0 + 16 * 1024, 1024, lane);
    v16h bn2 = load_b_f16(Bb + k0 + 32 * 1024, 1024, lane);
    v16h bn3 = load_b_f16(Bb + k0 + 48 * 1024, 1024, lane);
    c0 = WMMA_F16(a, b0, c0);
    c1 = WMMA_F16(a, b1, c1);
    c2 = WMMA_F16(a, b2, c2);
    c3 = WMMA_F16(a, b3, c3);
    a = an; b0 = bn0; b1 = bn1; b2 = bn2; b3 = bn3;
  }
  c0 = WMMA_F16(a, b0, c0);
  c1 = WMMA_F16(a, b1, c1);
  c2 = WMMA_F16(a, b2, c2);
  c3 = WMMA_F16(a, b3, c3);

  const int m0 = mt * 16 + ((lane >> 4) << 3);             // dim
  float br[8];
#pragma unroll
  for (int r = 0; r < 8; ++r) br[r] = bias[m0 + r];
  const int bidx = (tg * 64) >> 10;                        // group never crosses batch
  v8f cc[4] = {c0, c1, c2, c3};
#pragma unroll
  for (int j = 0; j < 4; ++j) {
    const int kpos = (tg * 64 + j * 16 + (lane & 15)) & 1023;
    _Float16* o = vt + ((size_t)bidx << 20) + (size_t)m0 * 1024 + kpos;
#pragma unroll
    for (int r = 0; r < 8; ++r)
      o[(size_t)r * 1024] = (_Float16)(cc[j][r] + br[r]);
  }
}

// ---------------------------------------------------------------------------
// K3: scores[bh,q,k] = (Q_h . K_h)/8 -> f32 into pos_attn region of d_out.
// grid(4,64,64) block(32,4): kg = bx*4+ty in 0..15 (4 k-tiles each), qt=by, bh=bz.
// ---------------------------------------------------------------------------
__global__ __launch_bounds__(128) void scores_k(const _Float16* __restrict__ Qh,
                                                const _Float16* __restrict__ Kh,
                                                float* __restrict__ attn) {
  const int lane = threadIdx.x;
  const int kg = blockIdx.x * blockDim.y + threadIdx.y;    // 0..15
  const int qt = blockIdx.y;                               // 0..63
  const int bh = blockIdx.z;                               // 0..63
  const int b = bh >> 4, h = bh & 15;
  const _Float16* Ab = Qh + (size_t)(b * 1024 + qt * 16) * 1024 + h * 64;
  const _Float16* Bb = Kh + (size_t)(b * 1024 + kg * 64) * 1024 + h * 64;
  v8f c0 = {}, c1 = {}, c2 = {}, c3 = {};
#pragma unroll
  for (int k0 = 0; k0 < 64; k0 += 32) {
    v16h a  = load_a_f16(Ab + k0, 1024, lane);
    v16h b0 = load_b_f16(Bb + k0,             1024, lane);
    v16h b1 = load_b_f16(Bb + k0 + 16 * 1024, 1024, lane);
    v16h b2 = load_b_f16(Bb + k0 + 32 * 1024, 1024, lane);
    v16h b3 = load_b_f16(Bb + k0 + 48 * 1024, 1024, lane);
    c0 = WMMA_F16(a, b0, c0);
    c1 = WMMA_F16(a, b1, c1);
    c2 = WMMA_F16(a, b2, c2);
    c3 = WMMA_F16(a, b3, c3);
  }
  float* o = attn + ((size_t)bh << 20) +
             (size_t)(qt * 16 + ((lane >> 4) << 3)) * 1024 + kg * 64 + (lane & 15);
  v8f cc[4] = {c0, c1, c2, c3};
#pragma unroll
  for (int j = 0; j < 4; ++j)
#pragma unroll
    for (int r = 0; r < 8; ++r)
      o[(size_t)r * 1024 + j * 16] = cc[j][r] * 0.125f;
}

// ---------------------------------------------------------------------------
// K4: softmax + neg-attention transform, one 256-thread block per row (in place).
// ---------------------------------------------------------------------------
__device__ __forceinline__ float blk_red_max(float v, float* red) {
  const int t = threadIdx.x;
  red[t] = v; __syncthreads();
  for (int s = 128; s > 0; s >>= 1) {
    if (t < s) red[t] = fmaxf(red[t], red[t + s]);
    __syncthreads();
  }
  float r = red[0]; __syncthreads();
  return r;
}
__device__ __forceinline__ float blk_red_sum(float v, float* red) {
  const int t = threadIdx.x;
  red[t] = v; __syncthreads();
  for (int s = 128; s > 0; s >>= 1) {
    if (t < s) red[t] += red[t + s];
    __syncthreads();
  }
  float r = red[0]; __syncthreads();
  return r;
}

__global__ __launch_bounds__(256) void softmax_negk(float* __restrict__ pos,
                                                    float* __restrict__ neg) {
  __shared__ float red[256];
  const size_t row = blockIdx.x;            // bh*1024 + q
  float* prow = pos + row * 1024;
  float* nrow = neg + row * 1024;
  const int t = threadIdx.x;

  float s[4];
#pragma unroll
  for (int j = 0; j < 4; ++j) s[j] = prow[t + j * 256];

  float m = fmaxf(fmaxf(s[0], s[1]), fmaxf(s[2], s[3]));
  m = blk_red_max(m, red);

  float e[4], psum = 0.f;
#pragma unroll
  for (int j = 0; j < 4; ++j) { e[j] = __expf(s[j] - m); psum += e[j]; }
  const float S = blk_red_sum(psum, red);
  const float invS = 1.0f / S;

  float p[4], nsum = 0.f;
#pragma unroll
  for (int j = 0; j < 4; ++j) { p[j] = e[j] * invS; nsum += 1.0f - p[j]; }
#pragma unroll
  for (int j = 0; j < 4; ++j) prow[t + j * 256] = p[j];

  const float t1 = blk_red_sum(nsum, red);
  const float inv1 = 1.0f / t1;

  float nc[4], nsum2 = 0.f;
#pragma unroll
  for (int j = 0; j < 4; ++j) {
    float v = (1.0f - p[j]) * inv1;
    float cap = 1e-6f / (p[j] + 1e-10f);
    v = fminf(v, cap);
    nc[j] = v; nsum2 += v;
  }
  const float t2 = blk_red_sum(nsum2, red);
  const float inv2 = 1.0f / t2;
#pragma unroll
  for (int j = 0; j < 4; ++j) nrow[t + j * 256] = nc[j] * inv2;
}

// ---------------------------------------------------------------------------
// K5: ctx[b,q,h*64+d] = sum_k attn[bh,q,k] * Vt[b,h*64+d,k]  (f16 out)
// One wave covers the whole head dim (4 tiles): attention tile read ONCE.
// grid(16,64) block(32,4): qt = bx*4+ty, bh = by.
// ---------------------------------------------------------------------------
__global__ __launch_bounds__(128) void attn_v(const float* __restrict__ attn,
                                              const _Float16* __restrict__ vt,
                                              _Float16* __restrict__ ctx) {
  const int lane = threadIdx.x;
  const int qt = blockIdx.x * blockDim.y + threadIdx.y;    // 0..63
  const int bh = blockIdx.y;                               // 0..63
  const int b = bh >> 4, h = bh & 15;
  const float* Ab = attn + ((size_t)bh << 20) + (size_t)(qt * 16) * 1024;
  const _Float16* Bb = vt + ((size_t)b << 20) + (size_t)(h * 64) * 1024;
  v8f c0 = {}, c1 = {}, c2 = {}, c3 = {};
  const int arow = (lane & 15) * 1024 + ((lane >> 4) << 3);

  v16h a  = load_a_f32(Ab, 1024, lane);
  v16h b0 = load_b_f16(Bb,             1024, lane);
  v16h b1 = load_b_f16(Bb + 16 * 1024, 1024, lane);
  v16h b2 = load_b_f16(Bb + 32 * 1024, 1024, lane);
  v16h b3 = load_b_f16(Bb + 48 * 1024, 1024, lane);
  for (int k0 = 32; k0 < 1024; k0 += 32) {
    // prefetch the DRAM-resident attention stream ~8 k-steps ahead
    if (k0 < 768) __builtin_prefetch((const void*)(Ab + arow + k0 + 256), 0, 0);
    v16h an  = load_a_f32(Ab + k0, 1024, lane);  // f32 attn -> f16 in-register
    v16h bn0 = load_b_f16(Bb + k0,             1024, lane);
    v16h bn1 = load_b_f16(Bb + k0 + 16 * 1024, 1024, lane);
    v16h bn2 = load_b_f16(Bb + k0 + 32 * 1024, 1024, lane);
    v16h bn3 = load_b_f16(Bb + k0 + 48 * 1024, 1024, lane);
    c0 = WMMA_F16(a, b0, c0);
    c1 = WMMA_F16(a, b1, c1);
    c2 = WMMA_F16(a, b2, c2);
    c3 = WMMA_F16(a, b3, c3);
    a = an; b0 = bn0; b1 = bn1; b2 = bn2; b3 = bn3;
  }
  c0 = WMMA_F16(a, b0, c0);
  c1 = WMMA_F16(a, b1, c1);
  c2 = WMMA_F16(a, b2, c2);
  c3 = WMMA_F16(a, b3, c3);

  const int q0 = qt * 16 + ((lane >> 4) << 3);
  _Float16* o = ctx + (size_t)(b * 1024 + q0) * 1024 + h * 64 + (lane & 15);
  v8f cc[4] = {c0, c1, c2, c3};
#pragma unroll
  for (int j = 0; j < 4; ++j)
#pragma unroll
    for (int r = 0; r < 8; ++r)
      o[(size_t)r * 1024 + j * 16] = (_Float16)cc[j][r];
}

// ---------------------------------------------------------------------------
// K6: out[m,n] = sum_k ctx[m,k]*Wo[n,k] + bo[n]  (f32 out into d_out)
// grid(256,4) block(32,4): wave = 16x64 strip.
// ---------------------------------------------------------------------------
__global__ __launch_bounds__(128) void out_proj(const _Float16* __restrict__ ctx,
                                                const _Float16* __restrict__ Wo,
                                                const float* __restrict__ bo,
                                                float* __restrict__ out) {
  const int lane = threadIdx.x;
  const int mt = blockIdx.x;                               // 0..255
  const int ng = blockIdx.y * blockDim.y + threadIdx.y;    // 0..15
  const _Float16* Ab = ctx + (size_t)(mt * 16) * 1024;
  const _Float16* Bb = Wo + (size_t)(ng * 64) * 1024;
  v8f c0 = {}, c1 = {}, c2 = {}, c3 = {};

  v16h a  = load_a_f16(Ab, 1024, lane);
  v16h b0 = load_b_f16(Bb,             1024, lane);
  v16h b1 = load_b_f16(Bb + 16 * 1024, 1024, lane);
  v16h b2 = load_b_f16(Bb + 32 * 1024, 1024, lane);
  v16h b3 = load_b_f16(Bb + 48 * 1024, 1024, lane);
  for (int k0 = 32; k0 < 1024; k0 += 32) {
    v16h an  = load_a_f16(Ab + k0, 1024, lane);
    v16h bn0 = load_b_f16(Bb + k0,             1024, lane);
    v16h bn1 = load_b_f16(Bb + k0 + 16 * 1024, 1024, lane);
    v16h bn2 = load_b_f16(Bb + k0 + 32 * 1024, 1024, lane);
    v16h bn3 = load_b_f16(Bb + k0 + 48 * 1024, 1024, lane);
    c0 = WMMA_F16(a, b0, c0);
    c1 = WMMA_F16(a, b1, c1);
    c2 = WMMA_F16(a, b2, c2);
    c3 = WMMA_F16(a, b3, c3);
    a = an; b0 = bn0; b1 = bn1; b2 = bn2; b3 = bn3;
  }
  c0 = WMMA_F16(a, b0, c0);
  c1 = WMMA_F16(a, b1, c1);
  c2 = WMMA_F16(a, b2, c2);
  c3 = WMMA_F16(a, b3, c3);

  const int nb = ng * 64 + (lane & 15);
  float* o = out + (size_t)(mt * 16 + ((lane >> 4) << 3)) * 1024 + nb;
  v8f cc[4] = {c0, c1, c2, c3};
#pragma unroll
  for (int j = 0; j < 4; ++j) {
    const float bn = bo[nb + j * 16];
#pragma unroll
    for (int r = 0; r < 8; ++r)
      o[(size_t)r * 1024 + j * 16] = cc[j][r] + bn;
  }
}

// ---------------------------------------------------------------------------
extern "C" void kernel_launch(void* const* d_in, const int* in_sizes, int n_in,
                              void* d_out, int out_size, void* d_ws, size_t ws_size,
                              hipStream_t stream) {
  (void)in_sizes; (void)n_in; (void)out_size; (void)ws_size;

  const float* query = (const float*)d_in[0];
  const float* key_  = (const float*)d_in[1];
  const float* value = (const float*)d_in[2];
  const float* Wq = (const float*)d_in[3];
  const float* bq = (const float*)d_in[4];
  const float* Wk = (const float*)d_in[5];
  const float* bk = (const float*)d_in[6];
  const float* Wv = (const float*)d_in[7];
  const float* bv = (const float*)d_in[8];
  const float* Wo = (const float*)d_in[9];
  const float* bo = (const float*)d_in[10];

  float* out = (float*)d_out;
  float* pos_out  = out;
  float* neg_out  = out + 4194304;     // 4*1024*1024
  float* pos_attn = out + 8388608;     // + 4*1024*1024
  float* neg_attn = out + 75497472;    // + 4*16*1024*1024

  char* ws = (char*)d_ws;
  _Float16* Xq16 = (_Float16*)(ws);                        // 8 MB
  _Float16* Xk16 = (_Float16*)(ws + ((size_t)8  << 20));   // 8 MB
  _Float16* Xv16 = (_Float16*)(ws + ((size_t)16 << 20));   // 8 MB
  _Float16* Wq16 = (_Float16*)(ws + ((size_t)24 << 20));   // 2 MB
  _Float16* Wk16 = (_Float16*)(ws + ((size_t)26 << 20));   // 2 MB
  _Float16* Wv16 = (_Float16*)(ws + ((size_t)28 << 20));   // 2 MB
  _Float16* Wo16 = (_Float16*)(ws + ((size_t)30 << 20));   // 2 MB
  _Float16* Qh   = (_Float16*)(ws + ((size_t)32 << 20));   // 8 MB
  _Float16* Kh   = (_Float16*)(ws + ((size_t)40 << 20));   // 8 MB
  _Float16* Vt   = (_Float16*)(ws + ((size_t)48 << 20));   // 8 MB
  _Float16* Cp   = (_Float16*)(ws + ((size_t)56 << 20));   // 8 MB
  _Float16* Cn   = (_Float16*)(ws + ((size_t)64 << 20));   // 8 MB

  const dim3 blk(32, 4);

  // One-time f32 -> f16 downconversion of activations and weights.
  cvt_f32_f16<<<dim3(4096), dim3(256), 0, stream>>>(query, Xq16, 1048576);
  cvt_f32_f16<<<dim3(4096), dim3(256), 0, stream>>>(key_,  Xk16, 1048576);
  cvt_f32_f16<<<dim3(4096), dim3(256), 0, stream>>>(value, Xv16, 1048576);
  cvt_f32_f16<<<dim3(1024), dim3(256), 0, stream>>>(Wq, Wq16, 262144);
  cvt_f32_f16<<<dim3(1024), dim3(256), 0, stream>>>(Wk, Wk16, 262144);
  cvt_f32_f16<<<dim3(1024), dim3(256), 0, stream>>>(Wv, Wv16, 262144);
  cvt_f32_f16<<<dim3(1024), dim3(256), 0, stream>>>(Wo, Wo16, 262144);

  // Projections (pure f16 WMMA inner loops, software-pipelined)
  proj_xwT<<<dim3(256, 4), blk, 0, stream>>>(Xq16, Wq16, bq, Qh);
  proj_xwT<<<dim3(256, 4), blk, 0, stream>>>(Xk16, Wk16, bk, Kh);
  proj_v_t<<<dim3(64, 16), blk, 0, stream>>>(Xv16, Wv16, bv, Vt);

  // Raw scores -> pos_attn region of d_out
  scores_k<<<dim3(4, 64, 64), blk, 0, stream>>>(Qh, Kh, pos_attn);

  // Softmax + neg-attention transform (in place on d_out)
  softmax_negk<<<dim3(65536), dim3(256), 0, stream>>>(pos_attn, neg_attn);

  // Attention-weighted values (pos and neg)
  attn_v<<<dim3(16, 64), blk, 0, stream>>>(pos_attn, Vt, Cp);
  attn_v<<<dim3(16, 64), blk, 0, stream>>>(neg_attn, Vt, Cn);

  // Output projections
  out_proj<<<dim3(256, 4), blk, 0, stream>>>(Cp, Wo16, bo, pos_out);
  out_proj<<<dim3(256, 4), blk, 0, stream>>>(Cn, Wo16, bo, neg_out);
}
